// GCNRegressor_9242769621112
// MI455X (gfx1250) — matrix operand order repaired and checked by reading, not compiled, not run on hardware
//
#include <hip/hip_runtime.h>

typedef __attribute__((ext_vector_type(16))) _Float16 v16h;
typedef __attribute__((ext_vector_type(8)))  float    v8f;

#define HIDC 128

// ---------------- small utility kernels ----------------

__global__ void fill_kernel(float* __restrict__ p, float v, int n) {
    int i = blockIdx.x * blockDim.x + threadIdx.x;
    if (i < n) p[i] = v;
}

__global__ void degree_kernel(const int* __restrict__ dst, float* __restrict__ deg, int nE) {
    int e = blockIdx.x * blockDim.x + threadIdx.x;
    if (e < nE) atomicAdd(&deg[dst[e]], 1.0f);
}

__global__ void dis_kernel(const float* __restrict__ deg, float* __restrict__ dis,
                           float* __restrict__ invdeg, int n) {
    int i = blockIdx.x * blockDim.x + threadIdx.x;
    if (i < n) {
        float d = deg[i];
        dis[i] = rsqrtf(d);
        invdeg[i] = 1.0f / d;
    }
}

__global__ void coef_kernel(const int* __restrict__ src, const int* __restrict__ dst,
                            const float* __restrict__ dis, float* __restrict__ coef, int nE) {
    int e = blockIdx.x * blockDim.x + threadIdx.x;
    if (e < nE) coef[e] = dis[src[e]] * dis[dst[e]];
}

// agg[i][c] = h[i][c] * invdeg[i] + bias[c]   (self-loop term + bias)
__global__ void agg_init_kernel(const float* __restrict__ h, const float* __restrict__ invdeg,
                                const float* __restrict__ bias, float* __restrict__ agg,
                                int nNodes) {
    int idx = blockIdx.x * blockDim.x + threadIdx.x;
    if (idx < nNodes * HIDC) {
        int i = idx >> 7;
        int c = idx & (HIDC - 1);
        agg[idx] = h[idx] * invdeg[i] + bias[c];
    }
}

// ---------------- WMMA GEMM: C[nrows x 128] = (relu?)(A[nrows x 128]) @ W[128 x 128] ----------------
// One workgroup = 8 waves = 128 rows; each wave owns a 16-row slab and the
// full 128 output columns (8 16x16 tiles), K swept in 4 chunks of 32.
// W is staged in LDS *pre-swizzled into B-fragment order*: Wl[chunk][tile][lane][16 halves],
// so each lane's B fragment is one contiguous 32-byte v16h (2x ds_load_b128)
// instead of 16 scalar ds_load_u16. f32 accumulation via v_wmma_f32_16x16x32_f16.
__global__ __launch_bounds__(256) void gemm_wmma_128(const float* __restrict__ A,
                                                     const float* __restrict__ Wg,
                                                     float* __restrict__ C,
                                                     int nrows, int relu_in) {
    // [4 k-chunks][8 col tiles][32 lanes][16 halves] = 16384 halves = 32 KB
    __shared__ _Float16 Wl[4 * 8 * 32 * 16];
    int tid = threadIdx.x;

    // Cooperative load + swizzle of W (64 KB f32 -> 32 KB f16, fragment order).
    // dest idx = ((c*8 + n)*32 + lane)*16 + e  maps to  W[c*32 + (lane>>4)*8 + (e&7) + (e>>3)*16][n*16 + (lane&15)]
    #pragma unroll 8
    for (int idx = tid; idx < 4 * 8 * 32 * 16; idx += 256) {
        int e  = idx & 15;
        int ln = (idx >> 4) & 31;
        int n  = (idx >> 9) & 7;
        int c  = idx >> 12;
        int k   = c * 32 + (ln >> 4) * 8 + (e & 7) + (e >> 3) * 16;
        int col = n * 16 + (ln & 15);
        Wl[idx] = (_Float16)Wg[k * HIDC + col];
    }
    __syncthreads();

    int wave   = tid >> 5;
    int lane   = tid & 31;
    int lane16 = lane & 15;
    int half   = lane >> 4;                 // 0 or 1
    int row0   = blockIdx.x * 128 + wave * 16;
    if (row0 >= nrows) return;              // nrows handled at 16-row tile granularity

    const float* arow = A + (size_t)(row0 + lane16) * HIDC;
    int ka = half * 8;                      // A: lane-half K offset within each 16-K group

    v8f zero = {0.f, 0.f, 0.f, 0.f, 0.f, 0.f, 0.f, 0.f};
    v8f acc[8];
    #pragma unroll
    for (int n = 0; n < 8; ++n) acc[n] = zero;

    #pragma unroll
    for (int c = 0; c < 4; ++c) {
        int kb = c * 32;
        // A fragment: e<8 -> K = kb + ka + e ; e>=8 -> K = kb + 16 + ka + (e-8)
        const float4* ar0 = (const float4*)(arow + kb + ka);        // 8 floats
        const float4* ar1 = (const float4*)(arow + kb + 16 + ka);   // 8 floats
        float4 a0 = ar0[0], a1 = ar0[1];
        float4 a2 = ar1[0], a3 = ar1[1];
        if (relu_in) {
            a0.x = fmaxf(a0.x, 0.f); a0.y = fmaxf(a0.y, 0.f); a0.z = fmaxf(a0.z, 0.f); a0.w = fmaxf(a0.w, 0.f);
            a1.x = fmaxf(a1.x, 0.f); a1.y = fmaxf(a1.y, 0.f); a1.z = fmaxf(a1.z, 0.f); a1.w = fmaxf(a1.w, 0.f);
            a2.x = fmaxf(a2.x, 0.f); a2.y = fmaxf(a2.y, 0.f); a2.z = fmaxf(a2.z, 0.f); a2.w = fmaxf(a2.w, 0.f);
            a3.x = fmaxf(a3.x, 0.f); a3.y = fmaxf(a3.y, 0.f); a3.z = fmaxf(a3.z, 0.f); a3.w = fmaxf(a3.w, 0.f);
        }
        v16h afrag;
        afrag[0]  = (_Float16)a0.x; afrag[1]  = (_Float16)a0.y;
        afrag[2]  = (_Float16)a0.z; afrag[3]  = (_Float16)a0.w;
        afrag[4]  = (_Float16)a1.x; afrag[5]  = (_Float16)a1.y;
        afrag[6]  = (_Float16)a1.z; afrag[7]  = (_Float16)a1.w;
        afrag[8]  = (_Float16)a2.x; afrag[9]  = (_Float16)a2.y;
        afrag[10] = (_Float16)a2.z; afrag[11] = (_Float16)a2.w;
        afrag[12] = (_Float16)a3.x; afrag[13] = (_Float16)a3.y;
        afrag[14] = (_Float16)a3.z; afrag[15] = (_Float16)a3.w;

        #pragma unroll
        for (int n = 0; n < 8; ++n) {
            // whole B fragment contiguous: one 32-byte vector LDS load
            v16h bfrag = *(const v16h*)&Wl[(((c * 8 + n) * 32) + lane) * 16];
            acc[n] = __builtin_amdgcn_wmma_f32_16x16x32_f16(
                false, afrag, false, bfrag, (short)0, acc[n], false, false);
        }
    }

    // D layout: element r -> row = row0 + 8*half + r, col = lane16 (+16*n)
    int rbase = row0 + half * 8;
    #pragma unroll
    for (int n = 0; n < 8; ++n) {
        int col = n * 16 + lane16;
        #pragma unroll
        for (int r = 0; r < 8; ++r)
            C[(size_t)(rbase + r) * HIDC + col] = acc[n][r];
    }
}

// ---------------- edge scatter: agg[dst] += h[src] * coef  (one wave32 per edge) ----------------
// h and agg are each 51.2 MB -> both L2-resident (192 MB L2); 512 B coalesced
// gather + 512 B of global_atomic_add_f32 scatter per edge.
__global__ __launch_bounds__(256) void edge_scatter_kernel(const int* __restrict__ src,
                                                           const int* __restrict__ dst,
                                                           const float* __restrict__ coef,
                                                           const float* __restrict__ h,
                                                           float* __restrict__ agg, int nE) {
    int gw   = (int)((blockIdx.x * 256u + threadIdx.x) >> 5);  // global wave id = edge id
    int lane = threadIdx.x & 31;
    if (gw >= nE) return;
    int   s = src[gw];
    int   d = dst[gw];
    float c = coef[gw];
    const float4* hv = (const float4*)(h + (size_t)s * HIDC);
    float4 v = hv[lane];
    float* a = agg + (size_t)d * HIDC + lane * 4;
    atomicAdd(a + 0, v.x * c);
    atomicAdd(a + 1, v.y * c);
    atomicAdd(a + 2, v.z * c);
    atomicAdd(a + 3, v.w * c);
}

// ---------------- pooling + FC head ----------------

__global__ void pool_accum_kernel(const float* __restrict__ agg, const int* __restrict__ batch,
                                  float* __restrict__ gsum, int nNodes) {
    int idx = blockIdx.x * blockDim.x + threadIdx.x;
    if (idx < nNodes * HIDC) {
        int i = idx >> 7;
        int c = idx & (HIDC - 1);
        float v = agg[idx];
        v = v > 0.0f ? v : 0.0f;  // relu before pooling
        atomicAdd(&gsum[(size_t)batch[i] * HIDC + c], v);
    }
}

__global__ void count_kernel(const int* __restrict__ batch, float* __restrict__ gcnt, int nNodes) {
    int i = blockIdx.x * blockDim.x + threadIdx.x;
    if (i < nNodes) atomicAdd(&gcnt[batch[i]], 1.0f);
}

__global__ void final_kernel(const float* __restrict__ gsum, const float* __restrict__ gcnt,
                             const float* __restrict__ Wfc, const float* __restrict__ bfc,
                             float* __restrict__ out) {
    __shared__ float red[HIDC];
    int g = blockIdx.x;
    int c = threadIdx.x;
    float cnt = gcnt[g];
    cnt = cnt > 1.0f ? cnt : 1.0f;
    red[c] = (gsum[(size_t)g * HIDC + c] / cnt) * Wfc[c];
    __syncthreads();
    #pragma unroll
    for (int sft = 64; sft > 0; sft >>= 1) {
        if (c < sft) red[c] += red[c + sft];
        __syncthreads();
    }
    if (c == 0) out[g] = red[0] + bfc[0];
}

// ---------------- orchestration ----------------

extern "C" void kernel_launch(void* const* d_in, const int* in_sizes, int n_in,
                              void* d_out, int out_size, void* d_ws, size_t ws_size,
                              hipStream_t stream) {
    const float* x          = (const float*)d_in[0];
    const int*   edge_index = (const int*)d_in[1];
    const int*   batch      = (const int*)d_in[2];
    const float* W1         = (const float*)d_in[3];
    const float* b1         = (const float*)d_in[4];
    const float* W2         = (const float*)d_in[5];
    const float* b2         = (const float*)d_in[6];
    const float* Wfc        = (const float*)d_in[7];
    const float* bfc        = (const float*)d_in[8];
    float*       out        = (float*)d_out;

    int nNodes  = in_sizes[2];
    int nE      = in_sizes[1] / 2;
    int nGraphs = out_size;
    const int* src  = edge_index;
    const int* dstI = edge_index + nE;

    // workspace carve-up (floats)
    float* ws     = (float*)d_ws;
    float* deg    = ws; ws += nNodes;
    float* dis    = ws; ws += nNodes;
    float* invdeg = ws; ws += nNodes;
    float* coef   = ws; ws += nE;
    float* H      = ws; ws += (size_t)nNodes * HIDC;   // linear output per layer
    float* AGG    = ws; ws += (size_t)nNodes * HIDC;   // aggregated output per layer
    float* gsum   = ws; ws += (size_t)nGraphs * HIDC;
    float* gcnt   = ws; ws += nGraphs;

    const int T = 256;
    int nElem = nNodes * HIDC;

    // degrees (with self-loop) and symmetric normalization
    fill_kernel<<<(nNodes + T - 1) / T, T, 0, stream>>>(deg, 1.0f, nNodes);
    degree_kernel<<<(nE + T - 1) / T, T, 0, stream>>>(dstI, deg, nE);
    dis_kernel<<<(nNodes + T - 1) / T, T, 0, stream>>>(deg, dis, invdeg, nNodes);
    coef_kernel<<<(nE + T - 1) / T, T, 0, stream>>>(src, dstI, dis, coef, nE);

    int gemmBlocks = (nNodes + 127) / 128;
    int edgeBlocks = (int)(((long long)nE * 32 + T - 1) / T);

    // ---- layer 1 ----
    gemm_wmma_128<<<gemmBlocks, T, 0, stream>>>(x, W1, H, nNodes, /*relu_in=*/0);
    agg_init_kernel<<<(nElem + T - 1) / T, T, 0, stream>>>(H, invdeg, b1, AGG, nNodes);
    edge_scatter_kernel<<<edgeBlocks, T, 0, stream>>>(src, dstI, coef, H, AGG, nE);

    // ---- layer 2 (relu of AGG fused into GEMM A-read) ----
    gemm_wmma_128<<<gemmBlocks, T, 0, stream>>>(AGG, W2, H, nNodes, /*relu_in=*/1);
    agg_init_kernel<<<(nElem + T - 1) / T, T, 0, stream>>>(H, invdeg, b2, AGG, nNodes);
    edge_scatter_kernel<<<edgeBlocks, T, 0, stream>>>(src, dstI, coef, H, AGG, nE);

    // ---- global mean pool (relu fused) + FC head ----
    fill_kernel<<<(nGraphs * HIDC + T - 1) / T, T, 0, stream>>>(gsum, 0.0f, nGraphs * HIDC);
    fill_kernel<<<(nGraphs + T - 1) / T, T, 0, stream>>>(gcnt, 0.0f, nGraphs);
    pool_accum_kernel<<<(nElem + T - 1) / T, T, 0, stream>>>(AGG, batch, gsum, nNodes);
    count_kernel<<<(nNodes + T - 1) / T, T, 0, stream>>>(batch, gcnt, nNodes);
    final_kernel<<<nGraphs, HIDC, 0, stream>>>(gsum, gcnt, Wfc, bfc, out);
}
